// MultiHeadAttention_15255723835908
// MI455X (gfx1250) — matrix operand order repaired and checked by reading, not compiled, not run on hardware
//
#include <hip/hip_runtime.h>
#include <cstdint>

typedef __attribute__((ext_vector_type(16))) __bf16 v16bf;
typedef __attribute__((ext_vector_type(8)))  float  v8f;
typedef unsigned short u16;

#define DEV __device__ __forceinline__

constexpr int S_LEN = 2048, D_MODEL = 1024, N_HEADS = 16, D_K = 64, BATCH = 4;
constexpr int M_TOTAL = BATCH * S_LEN;   // 8192 rows for the projection GEMMs

DEV u16 f2bf(float f) {                  // round-to-nearest-even fp32 -> bf16
  unsigned int u = __float_as_uint(f);
  u += 0x7FFFu + ((u >> 16) & 1u);
  return (u16)(u >> 16);
}

DEV v16bf load16(const u16* p0, const u16* p1) {  // two 16B chunks -> bf16x16 frag
  union { uint4 q[2]; v16bf v; } u;
  u.q[0] = *(const uint4*)p0;
  u.q[1] = *(const uint4*)p1;
  return u.v;
}

DEV v8f zero8() { v8f z;
#pragma unroll
  for (int i = 0; i < 8; ++i) z[i] = 0.0f;
  return z; }

DEV v8f wmma_bf16(v16bf a, v16bf b, v8f c) {
  return __builtin_amdgcn_wmma_f32_16x16x32_bf16(false, a, false, b, (short)0, c,
                                                 false, false);
}

// CDNA5 async global->LDS copy: per-lane 16B, tracked by ASYNCcnt (no VGPR data).
DEV void async_g2l_b128(unsigned lds_off, const void* gaddr) {
  asm volatile("global_load_async_to_lds_b128 %0, %1, off"
               :: "v"(lds_off), "v"((unsigned long long)(uintptr_t)gaddr)
               : "memory");
}
DEV void wait_async0() { asm volatile("s_wait_asynccnt 0x0" ::: "memory"); }

// ---------------------------------------------------------------- fp32 -> bf16
struct alignas(8) U16x4 { u16 v[4]; };
__global__ void cvt_bf16(const float4* __restrict__ in, U16x4* __restrict__ out,
                         int n4) {
  int i = blockIdx.x * blockDim.x + threadIdx.x;
  if (i < n4) {
    float4 f = in[i];
    U16x4 o;
    o.v[0] = f2bf(f.x); o.v[1] = f2bf(f.y); o.v[2] = f2bf(f.z); o.v[3] = f2bf(f.w);
    out[i] = o;
  }
}

// ------------------------------------------------- Y = X @ W^T + bias (bf16 WMMA)
// 128 threads = 4 waves (2x2), 64x64 tile per wave -> 16 WMMAs per 8 frag loads.
// MODE 0: bf16 head-split [B,H,S,Dk] out.  MODE 1: fp32 [M,N] out (d_out).
template <int MODE>
__global__ __launch_bounds__(128)
void gemm_xwT(const u16* __restrict__ A, const u16* __restrict__ W,
              const float* __restrict__ bias, u16* __restrict__ outBF,
              float* __restrict__ outF) {
  const int lane = threadIdx.x & 31, wave = threadIdx.x >> 5;
  const int l15 = lane & 15, hi = lane >> 4;
  const int wm = wave & 1, wn = wave >> 1;
  const int m0 = blockIdx.x * 128 + wm * 64;
  const int n0 = blockIdx.y * 128 + wn * 64;

  v8f acc[4][4];
#pragma unroll
  for (int i = 0; i < 4; ++i)
#pragma unroll
    for (int j = 0; j < 4; ++j) acc[i][j] = zero8();

  for (int k0 = 0; k0 < D_MODEL; k0 += 32) {
    v16bf a[4], b[4];
#pragma unroll
    for (int i = 0; i < 4; ++i) {   // A 16x32: lane=row, K chunks 0..7/16..23 (+8 if hi)
      const u16* rp = A + (size_t)(m0 + i * 16 + l15) * D_MODEL + k0 + hi * 8;
      a[i] = load16(rp, rp + 16);
    }
#pragma unroll
    for (int j = 0; j < 4; ++j) {   // B 32x16: lane=col(N)=W row, K chunks 0..7/16..23 (+8 if hi)
      const u16* rp = W + (size_t)(n0 + j * 16 + l15) * D_MODEL + k0 + hi * 8;
      b[j] = load16(rp, rp + 16);
    }
#pragma unroll
    for (int i = 0; i < 4; ++i)
#pragma unroll
      for (int j = 0; j < 4; ++j) acc[i][j] = wmma_bf16(a[i], b[j], acc[i][j]);
  }

  float bv[4];
#pragma unroll
  for (int j = 0; j < 4; ++j) bv[j] = bias[n0 + j * 16 + l15];

#pragma unroll
  for (int i = 0; i < 4; ++i)
#pragma unroll
    for (int j = 0; j < 4; ++j) {
      int n = n0 + j * 16 + l15;
#pragma unroll
      for (int r = 0; r < 8; ++r) {            // C layout: M = r + 8*hi, N = l15
        int m = m0 + i * 16 + r + 8 * hi;
        float val = acc[i][j][r] + bv[j];
        if (MODE == 0) {
          int bI = m >> 11, sI = m & 2047, hh = n >> 6, dd = n & 63;
          outBF[(((size_t)(bI * N_HEADS + hh) * S_LEN + sI) * D_K) + dd] = f2bf(val);
        } else {
          outF[(size_t)m * D_MODEL + n] = val;
        }
      }
    }
}

// -------------------------------------------- flash attention (causal), bf16 WMMA
// grid: (S/128, B*H).  8 waves/block, each wave owns 16 query rows.
// K tiles: async-DMA'd to LDS, double-buffered, prefetched one iteration ahead.
constexpr int KT_PITCH = 72;                 // halves; 144B row pitch (16B-aligned, bank-spread)
constexpr int KT_HALVES = 32 * KT_PITCH;     // one K tile buffer
constexpr int VT_PITCH = 40;
constexpr int VT_HALVES = 64 * VT_PITCH;

__global__ __launch_bounds__(256)
void attn_kernel(const u16* __restrict__ Q, const u16* __restrict__ K,
                 const u16* __restrict__ V, u16* __restrict__ ctx) {
  __shared__ u16 kt[2 * KT_HALVES];   // K tile (row-major, padded), double-buffered
  __shared__ u16 vt[2 * VT_HALVES];   // V tile transposed [d][k], double-buffered
  __shared__ u16 pbuf[8 * 16 * 32];   // per-wave P tile (C-layout -> A-layout relay)

  const int lane = threadIdx.x & 31, wave = threadIdx.x >> 5;
  const int l15 = lane & 15, hi = lane >> 4;
  const int bh = blockIdx.y;
  const int q0 = blockIdx.x * 128 + wave * 16;

  const u16* Qh = Q + (size_t)bh * S_LEN * D_K;
  const u16* Kh = K + (size_t)bh * S_LEN * D_K;
  const u16* Vh = V + (size_t)bh * S_LEN * D_K;

  const unsigned kt_base = (unsigned)(size_t)(void*)kt;   // LDS byte offset of kt
  const int t = threadIdx.x;
  const int krow = t >> 3, kchunk = t & 7;  // 256 threads x 16B = 32x64 bf16 tile
  const unsigned kt_lds = kt_base + (unsigned)(krow * KT_PITCH + kchunk * 8) * 2;

  v16bf aq[2];                        // Q rows as A fragments (Dk=64 -> 2 K-chunks)
#pragma unroll
  for (int c = 0; c < 2; ++c) {
    const u16* rp = Qh + (size_t)(q0 + l15) * D_K + c * 32 + hi * 8;
    aq[c] = load16(rp, rp + 16);
  }

  v8f o[4];
#pragma unroll
  for (int j = 0; j < 4; ++j) o[j] = zero8();
  float m_r[8], l_r[8];
#pragma unroll
  for (int r = 0; r < 8; ++r) { m_r[r] = -1e30f; l_r[r] = 0.0f; }

  const float scale = 0.125f;         // 1/sqrt(64)
  const int kend = (blockIdx.x + 1) * 128;

  // prologue: start async copy of first K tile into buffer 0
  async_g2l_b128(kt_lds, Kh + (size_t)krow * D_K + kchunk * 8);

  for (int kb = 0; kb < kend; kb += 32) {
    const int cur = (kb >> 5) & 1;
    u16* ktc = kt + cur * KT_HALVES;
    u16* vtc = vt + cur * VT_HALVES;

    {                                 // stage V[kb..kb+31][0..63] transposed into LDS
      union { uint4 q; u16 s[8]; } u;
      u.q = *(const uint4*)(Vh + (size_t)(kb + krow) * D_K + kchunk * 8);
      const int d0 = kchunk * 8;
#pragma unroll
      for (int j = 0; j < 8; ++j) vtc[(d0 + j) * VT_PITCH + krow] = u.s[j];
    }

    wait_async0();                    // K tile `cur` fully landed in LDS
    __syncthreads();                  // vt[cur]/kt[cur] ready; prior readers done

    if (kb + 32 < kend)               // prefetch next K tile; overlaps whole iteration
      async_g2l_b128(kt_lds + (unsigned)(cur ^ 1) * (KT_HALVES * 2),
                     Kh + (size_t)(kb + 32 + krow) * D_K + kchunk * 8);

    v8f sc[2];                        // scores 16x32 = two 16x16 C tiles
#pragma unroll
    for (int nt = 0; nt < 2; ++nt) {  // B col = K row, contiguous along Dk in LDS
      const u16* rp = ktc + (nt * 16 + l15) * KT_PITCH;
      v16bf bk0 = load16(rp + hi * 8, rp + hi * 8 + 16);
      v16bf bk1 = load16(rp + 32 + hi * 8, rp + 32 + hi * 8 + 16);
      v8f z = zero8();
      z = wmma_bf16(aq[0], bk0, z);
      z = wmma_bf16(aq[1], bk1, z);
      sc[nt] = z;
    }

    const bool needmask = (kb + 31) > q0;
#pragma unroll
    for (int nt = 0; nt < 2; ++nt)
#pragma unroll
      for (int r = 0; r < 8; ++r) {
        float v = sc[nt][r] * scale;
        if (needmask) {
          int col = kb + nt * 16 + l15;
          int row = q0 + r + 8 * hi;
          if (col > row) v = -1e30f;
        }
        sc[nt][r] = v;
      }

#pragma unroll
    for (int r = 0; r < 8; ++r) {     // online softmax per row (rows striped by half-wave)
      float cm = fmaxf(sc[0][r], sc[1][r]);
#pragma unroll
      for (int off = 8; off >= 1; off >>= 1) cm = fmaxf(cm, __shfl_xor(cm, off, 32));
      float mn = fmaxf(m_r[r], cm);
      float alpha = __expf(m_r[r] - mn);
      m_r[r] = mn;
      float p0 = __expf(sc[0][r] - mn);
      float p1 = __expf(sc[1][r] - mn);
      sc[0][r] = p0; sc[1][r] = p1;
      float rs = p0 + p1;
#pragma unroll
      for (int off = 8; off >= 1; off >>= 1) rs += __shfl_xor(rs, off, 32);
      l_r[r] = l_r[r] * alpha + rs;
#pragma unroll
      for (int j = 0; j < 4; ++j) o[j][r] *= alpha;
    }

    // P tile relay via wave-private LDS (same-wave DS ops are in-order: no barrier)
    u16* pw = pbuf + wave * 512;
#pragma unroll
    for (int nt = 0; nt < 2; ++nt)
#pragma unroll
      for (int r = 0; r < 8; ++r)
        pw[(r + 8 * hi) * 32 + nt * 16 + l15] = f2bf(sc[nt][r]);

    v16bf pa;                         // P as A fragment 16x32
    {
      const u16* rp = pw + l15 * 32 + hi * 8;
      pa = load16(rp, rp + 16);
    }
#pragma unroll
    for (int j = 0; j < 4; ++j) {     // V^T B fragments: contiguous along k in vt
      const u16* rp = vtc + (j * 16 + l15) * VT_PITCH + hi * 8;
      v16bf bv = load16(rp, rp + 16);
      o[j] = wmma_bf16(pa, bv, o[j]);
    }
  }

  // epilogue: ctx[b, s, h*64+d] as bf16 for the output projection
  const int b = bh >> 4, h = bh & 15;
#pragma unroll
  for (int r = 0; r < 8; ++r) {
    float inv = 1.0f / l_r[r];
    int row = q0 + r + 8 * hi;
    size_t base = ((size_t)b * S_LEN + row) * D_MODEL + h * D_K;
#pragma unroll
    for (int j = 0; j < 4; ++j)
      ctx[base + j * 16 + l15] = f2bf(o[j][r] * inv);
  }
}

// ---------------------------------------------------------------------- launch
extern "C" void kernel_launch(void* const* d_in, const int* in_sizes, int n_in,
                              void* d_out, int out_size, void* d_ws, size_t ws_size,
                              hipStream_t stream) {
  const float* q_in = (const float*)d_in[0];
  const float* k_in = (const float*)d_in[1];
  const float* v_in = (const float*)d_in[2];
  const float* w_q = (const float*)d_in[3];
  const float* b_q = (const float*)d_in[4];
  const float* w_k = (const float*)d_in[5];
  const float* b_k = (const float*)d_in[6];
  const float* w_v = (const float*)d_in[7];
  const float* b_v = (const float*)d_in[8];
  const float* w_o = (const float*)d_in[9];
  const float* b_o = (const float*)d_in[10];
  // d_in[11] = causal mask, hardcoded in attn_kernel

  char* ws = (char*)d_ws;
  const size_t XB = (size_t)M_TOTAL * D_MODEL * sizeof(u16);   // 16 MB
  const size_t WB = (size_t)D_MODEL * D_MODEL * sizeof(u16);   //  2 MB
  u16* Xq = (u16*)(ws);
  u16* Xk = (u16*)(ws + XB);
  u16* Xv = (u16*)(ws + 2 * XB);
  u16* Wq = (u16*)(ws + 3 * XB);
  u16* Wk = (u16*)(ws + 3 * XB + WB);
  u16* Wv = (u16*)(ws + 3 * XB + 2 * WB);
  u16* Wo = (u16*)(ws + 3 * XB + 3 * WB);
  u16* Qb = (u16*)(ws + 3 * XB + 4 * WB);
  u16* Kb = (u16*)(ws + 4 * XB + 4 * WB);
  u16* Vb = (u16*)(ws + 5 * XB + 4 * WB);
  u16* Cx = (u16*)(ws + 6 * XB + 4 * WB);

  const int nX4 = M_TOTAL * D_MODEL / 4;
  const int nW4 = D_MODEL * D_MODEL / 4;
  cvt_bf16<<<nX4 / 256, 256, 0, stream>>>((const float4*)q_in, (U16x4*)Xq, nX4);
  cvt_bf16<<<nX4 / 256, 256, 0, stream>>>((const float4*)k_in, (U16x4*)Xk, nX4);
  cvt_bf16<<<nX4 / 256, 256, 0, stream>>>((const float4*)v_in, (U16x4*)Xv, nX4);
  cvt_bf16<<<nW4 / 256, 256, 0, stream>>>((const float4*)w_q, (U16x4*)Wq, nW4);
  cvt_bf16<<<nW4 / 256, 256, 0, stream>>>((const float4*)w_k, (U16x4*)Wk, nW4);
  cvt_bf16<<<nW4 / 256, 256, 0, stream>>>((const float4*)w_v, (U16x4*)Wv, nW4);
  cvt_bf16<<<nW4 / 256, 256, 0, stream>>>((const float4*)w_o, (U16x4*)Wo, nW4);

  dim3 ggrid(M_TOTAL / 128, D_MODEL / 128);     // (64, 8)
  gemm_xwT<0><<<ggrid, 128, 0, stream>>>(Xq, Wq, b_q, Qb, nullptr);
  gemm_xwT<0><<<ggrid, 128, 0, stream>>>(Xk, Wk, b_k, Kb, nullptr);
  gemm_xwT<0><<<ggrid, 128, 0, stream>>>(Xv, Wv, b_v, Vb, nullptr);

  attn_kernel<<<dim3(S_LEN / 128, BATCH * N_HEADS), 256, 0, stream>>>(Qb, Kb, Vb, Cx);

  gemm_xwT<1><<<ggrid, 128, 0, stream>>>(Cx, Wo, b_o, nullptr, (float*)d_out);
}
